// GSPHAR_1348619731338
// MI455X (gfx1250) — matrix operand order repaired and checked by reading, not compiled, not run on hardware
//
#include <hip/hip_runtime.h>
#include <math.h>

#define Bb 64
#define Nn 512
#define NT (Nn / 16)

typedef __attribute__((ext_vector_type(16))) _Float16 v16h;
typedef __attribute__((ext_vector_type(8)))  _Float16 v8h;
typedef __attribute__((ext_vector_type(8)))  float    v8f;

// ---- 1. prep: normalized lag rows (f16, K padded to 32) + transposed X ------
__global__ void prep_kernel(const float* __restrict__ x1,
                            const float* __restrict__ x5,
                            const float* __restrict__ x22,
                            _Float16* __restrict__ z5,
                            _Float16* __restrict__ z22,
                            _Float16* __restrict__ Xt) {
  int idx = blockIdx.x * blockDim.x + threadIdx.x;
  if (idx >= Bb * Nn) return;
  int b = idx / Nn, n = idx % Nn;
  // lag5 normalized rows: corr = z.zT exactly ((L-1) factors cancel)
  {
    const float* x = x5 + (size_t)idx * 5;
    float mu = 0.f;
    for (int l = 0; l < 5; ++l) mu += x[l];
    mu *= 0.2f;
    float xc[5], ss = 0.f;
    for (int l = 0; l < 5; ++l) { xc[l] = x[l] - mu; ss += xc[l] * xc[l]; }
    float inv = ss > 1e-20f ? rsqrtf(ss) : 0.f;
    _Float16 tmp[32];
#pragma unroll
    for (int l = 0; l < 32; ++l) tmp[l] = (l < 5) ? (_Float16)(xc[l] * inv) : (_Float16)0.f;
    _Float16* z = z5 + (size_t)idx * 32;
#pragma unroll
    for (int l = 0; l < 32; ++l) z[l] = tmp[l];
  }
  // lag22 normalized rows
  {
    const float* x = x22 + (size_t)idx * 22;
    float mu = 0.f;
    for (int l = 0; l < 22; ++l) mu += x[l];
    mu *= (1.f / 22.f);
    float xc[22], ss = 0.f;
    for (int l = 0; l < 22; ++l) { xc[l] = x[l] - mu; ss += xc[l] * xc[l]; }
    float inv = ss > 1e-20f ? rsqrtf(ss) : 0.f;
    _Float16 tmp[32];
#pragma unroll
    for (int l = 0; l < 32; ++l) tmp[l] = (l < 22) ? (_Float16)(xc[l] * inv) : (_Float16)0.f;
    _Float16* z = z22 + (size_t)idx * 32;
#pragma unroll
    for (int l = 0; l < 32; ++l) z[l] = tmp[l];
  }
  // X transposed (lag-major) so spectral B fragments are contiguous in K:
  // Xt[b][lag][node], lag 0 = x_lag1, 1..5 = x_lag5, 6..27 = x_lag22, 28..31 = 0
  {
    _Float16* xt = Xt + ((size_t)b * 32) * Nn + n;
    xt[0] = (_Float16)x1[idx];
    const float* a = x5 + (size_t)idx * 5;
    for (int l = 0; l < 5; ++l) xt[(size_t)(1 + l) * Nn] = (_Float16)a[l];
    const float* c = x22 + (size_t)idx * 22;
    for (int l = 0; l < 22; ++l) xt[(size_t)(6 + l) * Nn] = (_Float16)c[l];
    for (int l = 28; l < 32; ++l) xt[(size_t)l * Nn] = (_Float16)0.f;
  }
}

// ---- 2. A_dir = relu(A - A^T) -----------------------------------------------
__global__ void adir_kernel(const float* __restrict__ A, float* __restrict__ Adir) {
  int idx = blockIdx.x * blockDim.x + threadIdx.x;
  if (idx >= Nn * Nn) return;
  int i = idx / Nn, j = idx % Nn;
  Adir[idx] = fmaxf(A[(size_t)i * Nn + j] - A[(size_t)j * Nn + i], 0.f);
}

// ---- WMMA fragment loaders (contiguous-K order, not the ISA table order, b128 vector loads) ----
// A (16-bit, 16x32): lane m=lane&15, halves = z[hi*16 .. +15] (contiguous)
__device__ __forceinline__ v16h load_a_frag(const _Float16* __restrict__ row, int hi) {
  return *(const v16h*)(row + hi * 16);
}
// B (16-bit, 32x16): lane n=lane&15, halves = z[hi*16 .. +15] (contiguous)
__device__ __forceinline__ v16h load_b_frag(const _Float16* __restrict__ row, int hi) {
  return *(const v16h*)(row + hi * 16);
}

// ---- 3. correlation Gram via WMMA: S = (0.5|Z5 Z5^T| + 0.5|Z22 Z22^T|)*Adir -
__global__ void __launch_bounds__(32)
corr_wmma_kernel(const _Float16* __restrict__ z5,
                 const _Float16* __restrict__ z22,
                 const float* __restrict__ Adir,
                 float* __restrict__ S) {
  int blk = blockIdx.x;                 // Bb * NT * NT blocks, one wave each
  int b   = blk / (NT * NT);
  int t   = blk % (NT * NT);
  int rowBase = (t / NT) * 16;
  int colBase = (t % NT) * 16;
  int lane = threadIdx.x & 31;
  int m = lane & 15, hi = lane >> 4;

  const _Float16* ar5  = z5  + ((size_t)(b * Nn + rowBase + m)) * 32;
  const _Float16* br5  = z5  + ((size_t)(b * Nn + colBase + m)) * 32;
  const _Float16* ar22 = z22 + ((size_t)(b * Nn + rowBase + m)) * 32;
  const _Float16* br22 = z22 + ((size_t)(b * Nn + colBase + m)) * 32;

  v16h a5  = load_a_frag(ar5,  hi);
  v16h b5  = load_b_frag(br5,  hi);
  v16h a22 = load_a_frag(ar22, hi);
  v16h b22 = load_b_frag(br22, hi);

  v8f acc5  = {};
  v8f acc22 = {};
  acc5  = __builtin_amdgcn_wmma_f32_16x16x32_f16(false, a5,  false, b5,  (short)0, acc5,  false, false);
  acc22 = __builtin_amdgcn_wmma_f32_16x16x32_f16(false, a22, false, b22, (short)0, acc22, false, false);

#pragma unroll
  for (int r = 0; r < 8; ++r) {
    int i = rowBase + r + 8 * hi;
    int j = colBase + m;
    float val = 0.5f * fabsf(acc5[r]) + 0.5f * fabsf(acc22[r]);
    S[((size_t)b * Nn + i) * Nn + j] = val * Adir[(size_t)i * Nn + j];
  }
}

// ---- 4. softmax over axis=1 (rows, per fixed column m) ----------------------
__global__ void softmax_kernel(float* __restrict__ S) {
  int idx = blockIdx.x * blockDim.x + threadIdx.x;
  if (idx >= Bb * Nn) return;
  int b = idx / Nn, m = idx % Nn;
  float* base = S + (size_t)b * Nn * Nn + m;
  float mx = -1e30f;
  for (int n = 0; n < Nn; ++n) mx = fmaxf(mx, base[(size_t)n * Nn]);
  float sum = 0.f;
  for (int n = 0; n < Nn; ++n) sum += __expf(base[(size_t)n * Nn] - mx);
  float rs = 1.f / sum;
  for (int n = 0; n < Nn; ++n) {
    float v = __expf(base[(size_t)n * Nn] - mx) * rs;
    base[(size_t)n * Nn] = v;
  }
}

// ---- 5. degree: dinv = rsqrt(sum_j A_s[i][j]) -------------------------------
__global__ void degree_kernel(const float* __restrict__ Ab, float* __restrict__ dinv) {
  int idx = blockIdx.x * blockDim.x + threadIdx.x;
  if (idx >= Bb * Nn) return;
  int b = idx / Nn, i = idx % Nn;
  const float* base = Ab + (size_t)b * Nn * Nn;
  float rs = 0.f, cs = 0.f;
  for (int j = 0; j < Nn; ++j) {
    rs += base[(size_t)i * Nn + j];
    cs += base[(size_t)j * Nn + i];
  }
  dinv[idx] = rsqrtf(0.5f * (rs + cs));
}

// ---- 6. magnetic Laplacian into work matrix M, V = I ------------------------
__global__ void lap_kernel(const float* __restrict__ Ab,
                           const float* __restrict__ dinv,
                           float2* __restrict__ Mw,
                           float2* __restrict__ Vv) {
  size_t idx = (size_t)blockIdx.x * blockDim.x + threadIdx.x;
  if (idx >= (size_t)Bb * Nn * Nn) return;
  int b = (int)(idx / ((size_t)Nn * Nn));
  int r = (int)(idx % ((size_t)Nn * Nn));
  int i = r / Nn, j = r % Nn;
  float aij = Ab[idx];
  float aji = Ab[((size_t)b * Nn + j) * Nn + i];
  float as  = 0.5f * (aij + aji);
  float th  = 1.5707963267948966f * (aij - aji);   // 2*pi*Q, Q=0.25
  float h   = dinv[b * Nn + i] * as * dinv[b * Nn + j];
  float sn, cs;
  __sincosf(th, &sn, &cs);
  float2 lv;
  lv.x = (i == j ? 1.f : 0.f) - h * cs;
  lv.y = -h * sn;
  Mw[idx] = lv;
  float2 vi; vi.x = (i == j ? 1.f : 0.f); vi.y = 0.f;
  Vv[idx] = vi;
}

// ---- 7. batched complex Hermitian Jacobi eigensolver ------------------------
// One workgroup (256 threads) per matrix; round-robin parallel ordering:
// 256 disjoint pairs per round, N-1 rounds per sweep.
__global__ void __launch_bounds__(256)
jacobi_kernel(float2* __restrict__ Mw, float2* __restrict__ Vv) {
  const int b = blockIdx.x;
  float2* M = Mw + (size_t)b * Nn * Nn;
  float2* V = Vv + (size_t)b * Nn * Nn;
  __shared__ int    pS[Nn / 2];
  __shared__ int    qS[Nn / 2];
  __shared__ float  cS[Nn / 2];
  __shared__ float2 sS[Nn / 2];
  const int t = threadIdx.x;

#pragma clang loop unroll(disable)
  for (int sweep = 0; sweep < 8; ++sweep) {
#pragma clang loop unroll(disable)
    for (int r = 0; r < Nn - 1; ++r) {
      // --- compute this round's rotation for pair t (tournament schedule) ---
      {
        int posA = t, posB = Nn - 1 - t;
        int p = (posA == 0) ? 0 : (1 + (posA - 1 + r) % (Nn - 1));
        int q = 1 + (posB - 1 + r) % (Nn - 1);
        if (p > q) { int tmp = p; p = q; q = tmp; }
        pS[t] = p; qS[t] = q;
        float  app = M[(size_t)p * Nn + p].x;
        float  aqq = M[(size_t)q * Nn + q].x;
        float2 apq = M[(size_t)p * Nn + q];
        float rmag = sqrtf(apq.x * apq.x + apq.y * apq.y);
        float c = 1.f;
        float2 s; s.x = 0.f; s.y = 0.f;
        if (rmag > 1e-12f) {
          float tau = (aqq - app) / (2.f * rmag);
          float tt  = (tau >= 0.f ? 1.f : -1.f) / (fabsf(tau) + sqrtf(1.f + tau * tau));
          c = rsqrtf(1.f + tt * tt);
          float sig = tt * c;
          float ir = sig / rmag;
          s.x = ir * apq.x;
          s.y = ir * apq.y;
        }
        cS[t] = c; sS[t] = s;
      }
      __syncthreads();
      // --- row phase: M <- J^H M (rows disjoint across pairs), float4-wide --
#pragma clang loop unroll(disable)
      for (int w = t; w < (Nn / 2) * (Nn / 2); w += 256) {
        int pr   = w >> 8;               // pair index
        int col2 = (w & 255) << 1;       // even column, 2 complex per iter
        int p2 = pS[pr], q2 = qS[pr];
        float  c2 = cS[pr];
        float2 s2 = sS[pr];
        float4* Pp = (float4*)(M + (size_t)p2 * Nn + col2);
        float4* Pq = (float4*)(M + (size_t)q2 * Nn + col2);
        float4 mp = *Pp;
        float4 mq = *Pq;
        float4 np, nq;
        // newrow_p = c*mp - s*mq ; newrow_q = conj(s)*mp + c*mq  (x2 complex)
        np.x = c2 * mp.x - (s2.x * mq.x - s2.y * mq.y);
        np.y = c2 * mp.y - (s2.x * mq.y + s2.y * mq.x);
        np.z = c2 * mp.z - (s2.x * mq.z - s2.y * mq.w);
        np.w = c2 * mp.w - (s2.x * mq.w + s2.y * mq.z);
        nq.x = c2 * mq.x + (s2.x * mp.x + s2.y * mp.y);
        nq.y = c2 * mq.y + (s2.x * mp.y - s2.y * mp.x);
        nq.z = c2 * mq.z + (s2.x * mp.z + s2.y * mp.w);
        nq.w = c2 * mq.w + (s2.x * mp.w - s2.y * mp.z);
        *Pp = np;
        *Pq = nq;
      }
      __syncthreads();
      // --- column phase: M <- M J, V <- V J (columns disjoint) --------------
#pragma clang loop unroll(disable)
      for (int w = t; w < (Nn / 2) * Nn; w += 256) {
        int pr  = w >> 9;
        int row = w & (Nn - 1);
        int p2 = pS[pr], q2 = qS[pr];
        float  c2 = cS[pr];
        float2 s2 = sS[pr];
        {
          float2 mp = M[(size_t)row * Nn + p2];
          float2 mq = M[(size_t)row * Nn + q2];
          float2 np, nq;
          // newcol_p = c*mp - conj(s)*mq ; newcol_q = s*mp + c*mq
          np.x = c2 * mp.x - (s2.x * mq.x + s2.y * mq.y);
          np.y = c2 * mp.y - (s2.x * mq.y - s2.y * mq.x);
          nq.x = c2 * mq.x + (s2.x * mp.x - s2.y * mp.y);
          nq.y = c2 * mq.y + (s2.x * mp.y + s2.y * mp.x);
          M[(size_t)row * Nn + p2] = np;
          M[(size_t)row * Nn + q2] = nq;
        }
        {
          float2 vp = V[(size_t)row * Nn + p2];
          float2 vq = V[(size_t)row * Nn + q2];
          float2 np, nq;
          np.x = c2 * vp.x - (s2.x * vq.x + s2.y * vq.y);
          np.y = c2 * vp.y - (s2.x * vq.y - s2.y * vq.x);
          nq.x = c2 * vq.x + (s2.x * vp.x - s2.y * vp.y);
          nq.y = c2 * vq.y + (s2.x * vp.y + s2.y * vp.x);
          V[(size_t)row * Nn + p2] = np;
          V[(size_t)row * Nn + q2] = nq;
        }
      }
      __syncthreads();
    }
  }
}

// ---- 8. spectral transform s = V @ X via WMMA (re and im planes) ------------
__global__ void __launch_bounds__(32)
spectral_wmma_kernel(const float2* __restrict__ Vv,
                     const _Float16* __restrict__ Xt,
                     float* __restrict__ sRe,
                     float* __restrict__ sIm) {
  int blk = blockIdx.x;                 // Bb * NT * 2 blocks
  int b = blk / (NT * 2);
  int t = blk % (NT * 2);
  int rowBase = (t >> 1) * 16;
  int colBase = (t & 1) * 16;
  int lane = threadIdx.x & 31;
  int m = lane & 15, hi = lane >> 4;

  v8f accR = {};
  v8f accI = {};
#pragma clang loop unroll(disable)
  for (int kb = 0; kb < Nn; kb += 32) {
    // A fragments: V row slice, two contiguous float2 runs of 8 -> float4 x4
    const float2* vrow = Vv + ((size_t)b * Nn + rowBase + m) * Nn + kb;
    const float4* run0 = (const float4*)(vrow + hi * 16);
    const float4* run1 = (const float4*)(vrow + hi * 16 + 8);
    v16h aR, aI;
#pragma unroll
    for (int j = 0; j < 4; ++j) {
      float4 e = run0[j];                 // halves (2j, 2j+1) of fragment
      aR[2 * j]     = (_Float16)e.x;
      aR[2 * j + 1] = (_Float16)e.z;
      aI[2 * j]     = (_Float16)e.y;
      aI[2 * j + 1] = (_Float16)e.w;
      float4 f = run1[j];                 // halves (8+2j, 8+2j+1)
      aR[8 + 2 * j]     = (_Float16)f.x;
      aR[8 + 2 * j + 1] = (_Float16)f.z;
      aI[8 + 2 * j]     = (_Float16)f.y;
      aI[8 + 2 * j + 1] = (_Float16)f.w;
    }
    // B fragment: transposed X -> contiguous 16 halves in K
    v16h bX = *(const v16h*)(Xt + ((size_t)(b * 32 + colBase + m)) * Nn + kb + hi * 16);

    accR = __builtin_amdgcn_wmma_f32_16x16x32_f16(false, aR, false, bX, (short)0, accR, false, false);
    accI = __builtin_amdgcn_wmma_f32_16x16x32_f16(false, aI, false, bX, (short)0, accI, false, false);
  }
#pragma unroll
  for (int r = 0; r < 8; ++r) {
    int row = rowBase + r + 8 * hi;
    int col = colBase + m;
    sRe[((size_t)b * Nn + row) * 32 + col] = accR[r];
    sIm[((size_t)b * Nn + row) * 32 + col] = accI[r];
  }
}

// ---- 9. lag mixing + complex linear -> y_spec -------------------------------
__global__ void yspec_kernel(const float* __restrict__ sRe,
                             const float* __restrict__ sIm,
                             const float* __restrict__ w5,
                             const float* __restrict__ w22,
                             const float* __restrict__ wr,
                             const float* __restrict__ br,
                             const float* __restrict__ wi,
                             const float* __restrict__ bi,
                             float2* __restrict__ ys) {
  int idx = blockIdx.x * blockDim.x + threadIdx.x;
  if (idx >= Bb * Nn) return;
  int n = idx % Nn;
  const float* sr = sRe + (size_t)idx * 32;
  const float* si = sIm + (size_t)idx * 32;
  float c5r = 0.f, c5i = 0.f;
  for (int l = 0; l < 5; ++l) {
    float w = w5[n * 5 + l];
    c5r += sr[1 + l] * w;
    c5i += si[1 + l] * w;
  }
  float c22r = 0.f, c22i = 0.f;
  for (int l = 0; l < 22; ++l) {
    float w = w22[n * 22 + l];
    c22r += sr[6 + l] * w;
    c22i += si[6 + l] * w;
  }
  float yr = wr[0] * sr[0] + wr[1] * c5r + wr[2] * c22r + br[0];
  float yi = wi[0] * si[0] + wi[1] * c5i + wi[2] * c22i + bi[0];
  float2 o; o.x = yr; o.y = yi;
  ys[idx] = o;
}

// ---- 10. y = V^H y_spec, then MLP 2->16->16->1 ------------------------------
__global__ void recon_mlp_kernel(const float2* __restrict__ Vv,
                                 const float2* __restrict__ ysAll,
                                 const float* __restrict__ W1, const float* __restrict__ b1,
                                 const float* __restrict__ W2, const float* __restrict__ b2,
                                 const float* __restrict__ W3, const float* __restrict__ b3,
                                 float* __restrict__ out) {
  int idx = blockIdx.x * blockDim.x + threadIdx.x;
  if (idx >= Bb * Nn) return;
  int b = idx / Nn, i = idx % Nn;
  const float2* Vb = Vv + (size_t)b * Nn * Nn;
  const float2* ys = ysAll + (size_t)b * Nn;
  float yr = 0.f, yi = 0.f;
  for (int j = 0; j < Nn; ++j) {
    if (j + 8 < Nn) __builtin_prefetch(&Vb[(size_t)(j + 8) * Nn + i], 0, 0);
    float2 v = Vb[(size_t)j * Nn + i];
    float2 s = ys[j];
    yr += v.x * s.x + v.y * s.y;   // conj(V) * ys
    yi += v.x * s.y - v.y * s.x;
  }
  float h1[16];
#pragma unroll
  for (int k = 0; k < 16; ++k)
    h1[k] = fmaxf(0.f, W1[k * 2] * yr + W1[k * 2 + 1] * yi + b1[k]);
  float h2[16];
#pragma unroll
  for (int k = 0; k < 16; ++k) {
    float a = b2[k];
#pragma unroll
    for (int j2 = 0; j2 < 16; ++j2) a += W2[k * 16 + j2] * h1[j2];
    h2[k] = fmaxf(0.f, a);
  }
  float o = b3[0];
#pragma unroll
  for (int j2 = 0; j2 < 16; ++j2) o += W3[j2] * h2[j2];
  out[idx] = fmaxf(0.f, o);
}

// ---- host launcher ----------------------------------------------------------
extern "C" void kernel_launch(void* const* d_in, const int* in_sizes, int n_in,
                              void* d_out, int out_size, void* d_ws, size_t ws_size,
                              hipStream_t stream) {
  const float* x1  = (const float*)d_in[0];
  const float* x5  = (const float*)d_in[1];
  const float* x22 = (const float*)d_in[2];
  const float* A   = (const float*)d_in[3];
  const float* w5  = (const float*)d_in[4];
  const float* w22 = (const float*)d_in[5];
  const float* W1  = (const float*)d_in[6];
  const float* b1  = (const float*)d_in[7];
  const float* W2  = (const float*)d_in[8];
  const float* b2  = (const float*)d_in[9];
  const float* W3  = (const float*)d_in[10];
  const float* b3  = (const float*)d_in[11];
  const float* wr  = (const float*)d_in[12];
  const float* br  = (const float*)d_in[13];
  const float* wi  = (const float*)d_in[14];
  const float* bi  = (const float*)d_in[15];
  float* out = (float*)d_out;

  char* ws = (char*)d_ws;
  size_t off = 0;
  auto alloc = [&](size_t bytes) -> char* {
    char* p = ws + off;
    off = (off + bytes + 255) & ~(size_t)255;
    return p;
  };
  _Float16* z5   = (_Float16*)alloc((size_t)Bb * Nn * 32 * 2);
  _Float16* z22  = (_Float16*)alloc((size_t)Bb * Nn * 32 * 2);
  _Float16* Xt   = (_Float16*)alloc((size_t)Bb * 32 * Nn * 2);
  float*    Adir = (float*)   alloc((size_t)Nn * Nn * 4);
  float*    S    = (float*)   alloc((size_t)Bb * Nn * Nn * 4);
  float*    dinv = (float*)   alloc((size_t)Bb * Nn * 4);
  float2*   Mw   = (float2*)  alloc((size_t)Bb * Nn * Nn * 8);
  float2*   Vv   = (float2*)  alloc((size_t)Bb * Nn * Nn * 8);
  float*    sRe  = (float*)   alloc((size_t)Bb * Nn * 32 * 4);
  float*    sIm  = (float*)   alloc((size_t)Bb * Nn * 32 * 4);
  float2*   ys   = (float2*)  alloc((size_t)Bb * Nn * 8);

  const int bn = Bb * Nn;                        // 32768
  const int bnn_blocks = (int)(((size_t)Bb * Nn * Nn + 255) / 256);

  prep_kernel<<<(bn + 255) / 256, 256, 0, stream>>>(x1, x5, x22, z5, z22, Xt);
  adir_kernel<<<(Nn * Nn + 255) / 256, 256, 0, stream>>>(A, Adir);
  corr_wmma_kernel<<<Bb * NT * NT, 32, 0, stream>>>(z5, z22, Adir, S);
  softmax_kernel<<<(bn + 255) / 256, 256, 0, stream>>>(S);
  degree_kernel<<<(bn + 255) / 256, 256, 0, stream>>>(S, dinv);
  lap_kernel<<<bnn_blocks, 256, 0, stream>>>(S, dinv, Mw, Vv);
  jacobi_kernel<<<Bb, 256, 0, stream>>>(Mw, Vv);
  spectral_wmma_kernel<<<Bb * NT * 2, 32, 0, stream>>>(Vv, Xt, sRe, sIm);
  yspec_kernel<<<(bn + 255) / 256, 256, 0, stream>>>(sRe, sIm, w5, w22, wr, br, wi, bi, ys);
  recon_mlp_kernel<<<(bn + 255) / 256, 256, 0, stream>>>(Vv, ys, W1, b1, W2, b2, W3, b3, out);
}
